// RWKV7_CrossAttention_78391743086645
// MI455X (gfx1250) — matrix operand order repaired and checked by reading, not compiled, not run on hardware
//
#include <hip/hip_runtime.h>
#include <stdint.h>

// ---------------------------------------------------------------------------
// RWKV7 cross-attention for MI455X (gfx1250, wave32, WMMA).
//
// Shapes: BATCH=4 QLEN=8 KVLEN=T=512 C=512 H=8 N=64; I = BATCH*QLEN = 32,
// BT = BATCH*T = 2048, IT = I*T = 16384.
//
// Algebraic reductions: kv-shift features are q-independent (2048 tokens for
// Wk/Wv/v-MLP/kk instead of 16384), w/a first stages split into kv-part
// (2048 tok) + q-part (32 tok), only t = T-1 needs r/g/groupnorm/Wo.
// GEMMs: v_wmma_f32_16x16x32_f16, A loaded as 2 contiguous b128 chunks per
// lane (row-major A maps exactly onto the ISA fragment layout), B weights
// pre-packed once into fragment order (1 contiguous 32B load per lane).
// Scan: 64x64 f32 state in registers (1 row/thread), operands staged via LDS,
// global_prefetch for the HBM-streamed tensors.
// ---------------------------------------------------------------------------

typedef _Float16 f16;
typedef __attribute__((ext_vector_type(16))) _Float16 v16h;
typedef __attribute__((ext_vector_type(8)))  _Float16 v8h;
typedef __attribute__((ext_vector_type(8)))  float    v8f;

#define BATCH 4
#define QLEN  8
#define TT    512
#define CC    512
#define HH    8
#define NN    64
#define II    32      // BATCH*QLEN
#define BT    2048    // BATCH*TT
#define IT    16384   // II*TT
#define GN_EPSF 6.4e-4f  // 1e-5 * 8^2

// ---------------- B-weight pre-pack into WMMA fragment order -----------------
// Bp[(kc*(N/16)+nt)*512 + lane*16 + e] = B[kc*32 + (e>>3)*16 + (lane>>4)*8 + (e&7)][nt*16 + (lane&15)]
__global__ void pack_b_k(const float* __restrict__ B, f16* __restrict__ Bp,
                         int K, int N) {
  int idx = blockIdx.x * 256 + threadIdx.x;
  if (idx >= K * N) return;
  int e    = idx & 15;
  int lane = (idx >> 4) & 31;
  int tile = idx >> 9;
  int ntiles = N >> 4;
  int kc = tile / ntiles, nt = tile - kc * ntiles;
  int k = kc * 32 + (e >> 3) * 16 + (lane >> 4) * 8 + (e & 7);
  int n = nt * 16 + (lane & 15);
  Bp[idx] = (f16)B[(size_t)k * N + n];
}

// ---------------- f16 WMMA GEMM: C[M,N](f32) = A[M,K] * B[K,N] ---------------
// A row-major f16 (contiguous per-lane chunks), B pre-packed (pack_b_k).
// One wave computes a 16 x (16*NT) strip; K multiple of 32.
template <int NT>
__global__ __launch_bounds__(32)
void wmma_gemm_f16(const f16* __restrict__ A, int lda,
                   const f16* __restrict__ Bp, int N,
                   float* __restrict__ C, int ldc, int K) {
  const int lane = threadIdx.x;
  const int row  = lane & 15;
  const int hi   = lane >> 4;
  const int nt0  = blockIdx.x * NT;     // first 16-col tile
  const int mt   = blockIdx.y * 16;
  const int ntiles = N >> 4;
  v8f acc[NT];
#pragma unroll
  for (int j = 0; j < NT; ++j) acc[j] = (v8f){};
  const f16* Arow = A + (size_t)(mt + row) * lda + hi * 8;
  for (int k0 = 0; k0 < K; k0 += 32) {
    // A(16x32) fragment: two contiguous 16B chunks per lane
    v8h alo = *(const v8h*)(Arow + k0);
    v8h ahi = *(const v8h*)(Arow + k0 + 16);
    v16h a;
#pragma unroll
    for (int e = 0; e < 8; ++e) { a[e] = alo[e]; a[8 + e] = ahi[e]; }
    const f16* bbase = Bp + ((size_t)(k0 >> 5) * ntiles + nt0) * 512 + lane * 16;
#pragma unroll
    for (int j = 0; j < NT; ++j) {
      v16h b = *(const v16h*)(bbase + j * 512);
      acc[j] = __builtin_amdgcn_wmma_f32_16x16x32_f16(false, a, false, b,
                                                      (short)0, acc[j], false, false);
    }
  }
#pragma unroll
  for (int j = 0; j < NT; ++j)
#pragma unroll
    for (int g = 0; g < 8; ++g)
      C[(size_t)(mt + g + hi * 8) * ldc + (nt0 + j) * 16 + row] = acc[j][g];
}

// ---------------- small elementwise helpers ---------------------------------
__global__ void f32_to_f16_k(const float* __restrict__ s, f16* __restrict__ d, int n) {
  int i = blockIdx.x * 256 + threadIdx.x;
  if (i < n) d[i] = (f16)s[i];
}

__global__ void sigmoid_to_f16_k(const float* __restrict__ s, f16* __restrict__ d, int n) {
  int i = blockIdx.x * 256 + threadIdx.x;
  if (i < n) d[i] = (f16)(1.f / (1.f + expf(-s[i])));
}

// buf <- sigmoid(bias[c] + buf), c = idx & 511
__global__ void epi_sigmoid_bias_k(float* __restrict__ buf, const float* __restrict__ bias, int n) {
  int i = blockIdx.x * 256 + threadIdx.x;
  if (i < n) {
    int c = i & (CC - 1);
    buf[i] = 1.f / (1.f + expf(-(bias[c] + buf[i])));
  }
}

// buf <- exp(-exp(-softplus(-(w0[c]+buf)) - 0.5))   (decay factor for scan)
__global__ void epi_wdecay_k(float* __restrict__ buf, const float* __restrict__ w0, int n) {
  int i = blockIdx.x * 256 + threadIdx.x;
  if (i < n) {
    int c = i & (CC - 1);
    float z = w0[c] + buf[i];
    float w = -log1pf(expf(-z)) - 0.5f;
    buf[i] = expf(-expf(w));
  }
}

// ---------------- token feature prep ----------------------------------------
__global__ __launch_bounds__(512)
void prep_tokens_k(const float* __restrict__ keyval,
                   const float* __restrict__ x_k, const float* __restrict__ x_v,
                   const float* __restrict__ x_w, const float* __restrict__ x_a,
                   f16* xk_h, f16* xv_h, f16* pkw_h, f16* pka_h) {
  int bt = blockIdx.x;            // b*512 + t
  int c  = threadIdx.x;
  int t  = bt & (TT - 1);
  float kv  = keyval[(size_t)bt * CC + c];
  float kvp = (t == 0) ? 0.f : keyval[(size_t)(bt - 1) * CC + c];
  float dx  = kvp - kv;
  size_t o = (size_t)bt * CC + c;
  xk_h[o]  = (f16)(kv + dx * x_k[c]);
  xv_h[o]  = (f16)(kv + dx * x_v[c]);
  pkw_h[o] = (f16)(kv * (1.f - x_w[c]));
  pka_h[o] = (f16)(kv * (1.f - x_a[c]));
}

// per i (t = T-1 only): xr, xg (cross mix, f16)
__global__ __launch_bounds__(512)
void prep_last_k(const float* __restrict__ query, const float* __restrict__ keyval,
                 const float* __restrict__ x_r, const float* __restrict__ x_g,
                 f16* xr_h, f16* xg_h) {
  int i = blockIdx.x, c = threadIdx.x;
  int b = i >> 3;
  float kv = keyval[((size_t)b * TT + (TT - 1)) * CC + c];
  float q  = query[(size_t)i * CC + c];
  float d  = q - kv;
  xr_h[(size_t)i * CC + c] = (f16)(kv + d * x_r[c]);
  xg_h[(size_t)i * CC + c] = (f16)(kv + d * x_g[c]);
}

// q-part of the w/a first stages: Q[i,d] = sum_c q[i,c]*xm[c]*W1[c,d]
__global__ __launch_bounds__(32)
void q_stage1_k(const float* __restrict__ query,
                const float* __restrict__ x_w, const float* __restrict__ x_a,
                const float* __restrict__ w1, const float* __restrict__ a1,
                float* Qw, float* Qa) {
  int i = blockIdx.x, which = blockIdx.y, d = threadIdx.x;
  const float* xm = which ? x_a : x_w;
  const float* W  = which ? a1 : w1;
  float* Q        = which ? Qa : Qw;
  const float* q = query + (size_t)i * CC;
  float s = 0.f;
  for (int c = 0; c < CC; ++c) s += q[c] * xm[c] * W[c * 32 + d];
  Q[i * 32 + d] = s;
}

// combine kv-part + q-part of first stage; tanh for w path, identity for a
__global__ void combine_hidden_k(const float* __restrict__ Pw, const float* __restrict__ Pa,
                                 const float* __restrict__ Qw, const float* __restrict__ Qa,
                                 f16* hw_h, f16* ha_h) {
  int idx = blockIdx.x * 256 + threadIdx.x;      // over IT*32
  if (idx >= IT * 32) return;
  int d  = idx & 31;
  int it = idx >> 5;
  int t  = it & (TT - 1);
  int i  = it >> 9;
  int bt = (i >> 3) * TT + t;
  hw_h[idx] = (f16)tanhf(Pw[bt * 32 + d] + Qw[i * 32 + d]);
  ha_h[idx] = (f16)(Pa[bt * 32 + d] + Qa[i * 32 + d]);
}

// kk = per-head L2-normalized (k0 * k_k)
__global__ __launch_bounds__(512)
void post_k_k(const float* __restrict__ k0, const float* __restrict__ k_k,
              float* __restrict__ kk) {
  __shared__ float red[CC];
  int bt = blockIdx.x, c = threadIdx.x;
  int h = c >> 6, l = c & 63;
  float v = k0[(size_t)bt * CC + c] * k_k[c];
  red[c] = v * v;
  __syncthreads();
  for (int off = 32; off >= 1; off >>= 1) {
    if (l < off) red[c] += red[c + off];
    __syncthreads();
  }
  float nrm = sqrtf(red[h * 64]);
  kk[(size_t)bt * CC + c] = v / fmaxf(nrm, 1e-12f);
}

// ---------------- the sequential RWKV7 scan ---------------------------------
// grid (I=32, H=8), 64 threads: thread r owns state row r (64 f32 regs).
// state[r,j] = state[r,j]*w[j] + (state[r,:]·a)*b[j] + v[r]*k[j]
__global__ __launch_bounds__(64)
void rwkv_scan_k(const float* __restrict__ wdec, const float* __restrict__ asig,
                 const float* __restrict__ kk, const float* __restrict__ k0,
                 const float* __restrict__ vbase, const float* __restrict__ sigv,
                 const float* __restrict__ v_first, const float* __restrict__ k_a,
                 const float* __restrict__ r_last, float* __restrict__ out_last) {
  const int i = blockIdx.x, h = blockIdx.y, tid = threadIdx.x;
  const int b = i >> 3;
  const int c = h * NN + tid;
  const float ka_c = k_a[c];
  float s[NN];
#pragma unroll
  for (int j = 0; j < NN; ++j) s[j] = 0.f;
  __shared__ float4 tab[NN];     // {w, k_final, a(-kk), b(kk*asig)}
  for (int t = 0; t < TT; ++t) {
    const size_t it = (size_t)i * TT + t;
    const size_t bt = (size_t)b * TT + t;
    // prefetch the HBM-streamed operands a few steps ahead (global_prefetch_b8)
    if (t + 8 < TT) {
      __builtin_prefetch(&wdec[(it + 8) * CC + c], 0, 1);
      __builtin_prefetch(&asig[(it + 8) * CC + c], 0, 1);
      __builtin_prefetch(&v_first[(it + 8) * CC + c], 0, 1);
    }
    float w_c = wdec[it * CC + c];
    float a_c = asig[it * CC + c];
    float kkc = kk[bt * CC + c];
    float k0c = k0[bt * CC + c];
    float vb  = vbase[bt * CC + c];
    float sg  = sigv[bt * CC + c];
    float vf  = v_first[it * CC + c];
    float kf  = k0c * (1.f + (a_c - 1.f) * ka_c);
    float vr  = vb + (vf - vb) * sg;       // this thread's row value
    __syncthreads();
    tab[tid] = make_float4(w_c, kf, -kkc, kkc * a_c);
    __syncthreads();
    float sa = 0.f;
#pragma unroll
    for (int j = 0; j < NN; ++j) sa += s[j] * tab[j].z;
#pragma unroll
    for (int j = 0; j < NN; ++j) {
      float4 e = tab[j];
      s[j] = s[j] * e.x + sa * e.w + vr * e.y;
    }
  }
  // out[i] = state · r  (only t = T-1 is consumed downstream)
  __syncthreads();
  __shared__ float rsh[NN];
  rsh[tid] = r_last[(size_t)i * CC + c];
  __syncthreads();
  float o = 0.f;
#pragma unroll
  for (int j = 0; j < NN; ++j) o += s[j] * rsh[j];
  out_last[(size_t)i * CC + c] = o;
}

// ---------------- last-token epilogue: groupnorm + rk*v + gate --------------
__global__ __launch_bounds__(512)
void finalize_k(const float* __restrict__ out_last, const float* __restrict__ r_last,
                const float* __restrict__ k0, const float* __restrict__ asig,
                const float* __restrict__ k_a, const float* __restrict__ vbase,
                const float* __restrict__ sigv, const float* __restrict__ v_first,
                const float* __restrict__ r_k, const float* __restrict__ ln_w,
                const float* __restrict__ ln_b, const float* __restrict__ g_last,
                f16* __restrict__ xo_h) {
  __shared__ float red[CC], red2[CC];
  int i = blockIdx.x, c = threadIdx.x;
  int h = c >> 6, l = c & 63;
  int b = i >> 3;
  const size_t bt = (size_t)b * TT + (TT - 1);
  const size_t it = (size_t)i * TT + (TT - 1);
  float y = out_last[(size_t)i * CC + c];
  red[c] = y; red2[c] = y * y;
  __syncthreads();
  for (int off = 32; off >= 1; off >>= 1) {
    if (l < off) { red[c] += red[c + off]; red2[c] += red2[c + off]; }
    __syncthreads();
  }
  float mu  = red[h * 64] * (1.f / 64.f);
  float var = red2[h * 64] * (1.f / 64.f) - mu * mu;
  float xo  = (y - mu) * rsqrtf(var + GN_EPSF) * ln_w[c] + ln_b[c];
  float a_c = asig[it * CC + c];
  float kf  = k0[bt * CC + c] * (1.f + (a_c - 1.f) * k_a[c]);
  float r   = r_last[(size_t)i * CC + c];
  __syncthreads();
  red[c] = r * kf * r_k[c];
  __syncthreads();
  for (int off = 32; off >= 1; off >>= 1) {
    if (l < off) red[c] += red[c + off];
    __syncthreads();
  }
  float rk = red[h * 64];
  float vb = vbase[bt * CC + c], sg = sigv[bt * CC + c], vf = v_first[it * CC + c];
  float v  = vb + (vf - vb) * sg;
  xo = (xo + rk * v) * g_last[(size_t)i * CC + c];
  xo_h[(size_t)i * CC + c] = (f16)xo;
}

// ---------------------------------------------------------------------------
static inline int ceil_div(int a, int b) { return (a + b - 1) / b; }

extern "C" void kernel_launch(void* const* d_in, const int* in_sizes, int n_in,
                              void* d_out, int out_size, void* d_ws, size_t ws_size,
                              hipStream_t stream) {
  (void)in_sizes; (void)n_in; (void)out_size; (void)ws_size;
  const float* query   = (const float*)d_in[0];
  const float* keyval  = (const float*)d_in[1];
  const float* v_first = (const float*)d_in[2];
  const float* x_r = (const float*)d_in[3];
  const float* x_w = (const float*)d_in[4];
  const float* x_k = (const float*)d_in[5];
  const float* x_v = (const float*)d_in[6];
  const float* x_a = (const float*)d_in[7];
  const float* x_g = (const float*)d_in[8];
  const float* w0  = (const float*)d_in[9];
  const float* w1  = (const float*)d_in[10];
  const float* w2  = (const float*)d_in[11];
  const float* a0  = (const float*)d_in[12];
  const float* a1  = (const float*)d_in[13];
  const float* a2  = (const float*)d_in[14];
  const float* v0  = (const float*)d_in[15];
  const float* v1  = (const float*)d_in[16];
  const float* v2  = (const float*)d_in[17];
  const float* g1  = (const float*)d_in[18];
  const float* g2  = (const float*)d_in[19];
  const float* k_k = (const float*)d_in[20];
  const float* k_a = (const float*)d_in[21];
  const float* r_k = (const float*)d_in[22];
  const float* Wr  = (const float*)d_in[23];
  const float* Wk  = (const float*)d_in[24];
  const float* Wv  = (const float*)d_in[25];
  const float* Wo  = (const float*)d_in[26];
  const float* ln_w = (const float*)d_in[27];
  const float* ln_b = (const float*)d_in[28];
  float* out = (float*)d_out;

  // -------- workspace carve-up (~97 MB total) -------------------------------
  uint8_t* base = (uint8_t*)d_ws;
  size_t off = 0;
  auto take = [&](size_t bytes) -> void* {
    void* p = base + off;
    off = (off + bytes + 255) & ~(size_t)255;
    return p;
  };
  f16* xk_h  = (f16*)take((size_t)BT * CC * 2);
  f16* xv_h  = (f16*)take((size_t)BT * CC * 2);
  f16* pkw_h = (f16*)take((size_t)BT * CC * 2);
  f16* pka_h = (f16*)take((size_t)BT * CC * 2);
  f16* xr_h  = (f16*)take((size_t)II * CC * 2);
  f16* xg_h  = (f16*)take((size_t)II * CC * 2);
  f16* xo_h  = (f16*)take((size_t)II * CC * 2);
  f16* Wk_p  = (f16*)take((size_t)CC * CC * 2);     // fragment-packed weights
  f16* Wv_p  = (f16*)take((size_t)CC * CC * 2);
  f16* Wr_p  = (f16*)take((size_t)CC * CC * 2);
  f16* Wo_p  = (f16*)take((size_t)CC * CC * 2);
  f16* w1_p  = (f16*)take((size_t)CC * 32 * 2);
  f16* a1_p  = (f16*)take((size_t)CC * 32 * 2);
  f16* v1_p  = (f16*)take((size_t)CC * 32 * 2);
  f16* w2_p  = (f16*)take((size_t)32 * CC * 2);
  f16* a2_p  = (f16*)take((size_t)32 * CC * 2);
  f16* v2_p  = (f16*)take((size_t)32 * CC * 2);
  f16* g1_p  = (f16*)take((size_t)CC * 96 * 2);
  f16* g2_p  = (f16*)take((size_t)96 * CC * 2);
  float* k0    = (float*)take((size_t)BT * CC * 4);
  float* vbase = (float*)take((size_t)BT * CC * 4);
  float* kk    = (float*)take((size_t)BT * CC * 4);
  float* sigv  = (float*)take((size_t)BT * CC * 4);   // GEMM out, sigmoid in place
  float* hv32  = (float*)take((size_t)BT * 32 * 4);
  f16*   hv_h  = (f16*)take((size_t)BT * 32 * 2);
  float* Pw32  = (float*)take((size_t)BT * 32 * 4);
  float* Pa32  = (float*)take((size_t)BT * 32 * 4);
  float* Qw32  = (float*)take((size_t)II * 32 * 4);
  float* Qa32  = (float*)take((size_t)II * 32 * 4);
  f16*   hw_h  = (f16*)take((size_t)IT * 32 * 2);
  f16*   ha_h  = (f16*)take((size_t)IT * 32 * 2);
  float* wdec  = (float*)take((size_t)IT * CC * 4);   // GEMM out, decay in place
  float* asig  = (float*)take((size_t)IT * CC * 4);   // GEMM out, sigmoid in place
  float* hg32  = (float*)take((size_t)II * 96 * 4);
  f16*   hg_h  = (f16*)take((size_t)II * 96 * 2);
  float* r_last   = (float*)take((size_t)II * CC * 4);
  float* g_last   = (float*)take((size_t)II * CC * 4);
  float* out_last = (float*)take((size_t)II * CC * 4);

  // -------- 1) pack B-side weights into WMMA fragment order -----------------
  struct { const float* s; f16* d; int K, N; } pk[] = {
    {Wk, Wk_p, CC, CC}, {Wv, Wv_p, CC, CC}, {Wr, Wr_p, CC, CC}, {Wo, Wo_p, CC, CC},
    {w1, w1_p, CC, 32}, {a1, a1_p, CC, 32}, {v1, v1_p, CC, 32},
    {w2, w2_p, 32, CC}, {a2, a2_p, 32, CC}, {v2, v2_p, 32, CC},
    {g1, g1_p, CC, 96}, {g2, g2_p, 96, CC},
  };
  for (auto& p : pk)
    pack_b_k<<<ceil_div(p.K * p.N, 256), 256, 0, stream>>>(p.s, p.d, p.K, p.N);

  // -------- 2) token feature prep -------------------------------------------
  prep_tokens_k<<<BT, CC, 0, stream>>>(keyval, x_k, x_v, x_w, x_a,
                                       xk_h, xv_h, pkw_h, pka_h);
  prep_last_k<<<II, CC, 0, stream>>>(query, keyval, x_r, x_g, xr_h, xg_h);
  q_stage1_k<<<dim3(II, 2), 32, 0, stream>>>(query, x_w, x_a, w1, a1, Qw32, Qa32);

  // -------- 3) big WMMA GEMMs -----------------------------------------------
  wmma_gemm_f16<4><<<dim3(8, BT / 16), 32, 0, stream>>>(xk_h, CC, Wk_p, CC, k0, CC, CC);
  wmma_gemm_f16<4><<<dim3(8, BT / 16), 32, 0, stream>>>(xv_h, CC, Wv_p, CC, vbase, CC, CC);
  wmma_gemm_f16<2><<<dim3(1, BT / 16), 32, 0, stream>>>(xv_h, CC, v1_p, 32, hv32, 32, CC);
  wmma_gemm_f16<2><<<dim3(1, BT / 16), 32, 0, stream>>>(pkw_h, CC, w1_p, 32, Pw32, 32, CC);
  wmma_gemm_f16<2><<<dim3(1, BT / 16), 32, 0, stream>>>(pka_h, CC, a1_p, 32, Pa32, 32, CC);
  wmma_gemm_f16<4><<<dim3(8, II / 16), 32, 0, stream>>>(xr_h, CC, Wr_p, CC, r_last, CC, CC);
  wmma_gemm_f16<2><<<dim3(3, II / 16), 32, 0, stream>>>(xg_h, CC, g1_p, 96, hg32, 96, CC);

  // -------- 4) small epilogues / second stages ------------------------------
  post_k_k<<<BT, CC, 0, stream>>>(k0, k_k, kk);
  f32_to_f16_k<<<ceil_div(BT * 32, 256), 256, 0, stream>>>(hv32, hv_h, BT * 32);
  sigmoid_to_f16_k<<<ceil_div(II * 96, 256), 256, 0, stream>>>(hg32, hg_h, II * 96);
  combine_hidden_k<<<ceil_div(IT * 32, 256), 256, 0, stream>>>(Pw32, Pa32, Qw32, Qa32,
                                                               hw_h, ha_h);
  // v-mix sigmoid over 2048 tokens
  wmma_gemm_f16<4><<<dim3(8, BT / 16), 32, 0, stream>>>(hv_h, 32, v2_p, CC, sigv, CC, 32);
  epi_sigmoid_bias_k<<<ceil_div(BT * CC, 256), 256, 0, stream>>>(sigv, v0, BT * CC);
  // w decay + a sigmoid over all 16384 tokens
  wmma_gemm_f16<4><<<dim3(8, IT / 16), 32, 0, stream>>>(hw_h, 32, w2_p, CC, wdec, CC, 32);
  epi_wdecay_k<<<ceil_div(IT * CC, 256), 256, 0, stream>>>(wdec, w0, IT * CC);
  wmma_gemm_f16<4><<<dim3(8, IT / 16), 32, 0, stream>>>(ha_h, 32, a2_p, CC, asig, CC, 32);
  epi_sigmoid_bias_k<<<ceil_div(IT * CC, 256), 256, 0, stream>>>(asig, a0, IT * CC);
  // gate g (32 tokens)
  wmma_gemm_f16<4><<<dim3(8, II / 16), 32, 0, stream>>>(hg_h, 96, g2_p, CC, g_last, CC, 96);

  // -------- 5) sequential scan ----------------------------------------------
  rwkv_scan_k<<<dim3(II, HH), NN, 0, stream>>>(wdec, asig, kk, k0, vbase, sigv,
                                               v_first, k_a, r_last, out_last);

  // -------- 6) last-token epilogue + Wo GEMM --------------------------------
  finalize_k<<<II, CC, 0, stream>>>(out_last, r_last, k0, asig, k_a, vbase, sigv,
                                    v_first, r_k, ln_w, ln_b, g_last, xo_h);
  wmma_gemm_f16<4><<<dim3(8, II / 16), 32, 0, stream>>>(xo_h, CC, Wo_p, CC, out, CC, CC);

  // -------- 7) second tuple output: v_first passthrough ---------------------
  hipMemcpyAsync(out + (size_t)II * CC, v_first,
                 (size_t)IT * CC * sizeof(float), hipMemcpyDeviceToDevice, stream);
}
